// SequenceRNNDecoder_78563541778993
// MI455X (gfx1250) — matrix operand order repaired and checked by reading, not compiled, not run on hardware
//
#include <hip/hip_runtime.h>
#include <math.h>

// Problem dims (fixed by the reference)
constexpr int BB  = 32;     // batch
constexpr int TT  = 128;    // time steps
constexpr int HH  = 512;    // hidden
constexpr int VV  = 32000;  // vocab
constexpr int G3H = 3 * HH; // 1536

// CDNA5 WMMA types (wave32)
typedef __attribute__((ext_vector_type(16))) __bf16 v16bf;
typedef __attribute__((ext_vector_type(8)))  __bf16 v8bf;
typedef __attribute__((ext_vector_type(8)))  float  v8f;
typedef __attribute__((ext_vector_type(4)))  float  v4f;
typedef __attribute__((ext_vector_type(8)))  unsigned short v8us;

// ---------------- device helpers ----------------

__device__ inline unsigned short f2bf(float f) {
  union { float f; unsigned int u; } v; v.f = f;
  unsigned int r = v.u + 0x7FFFu + ((v.u >> 16) & 1u); // round-to-nearest-even
  return (unsigned short)(r >> 16);
}

// A-fragment (16x32 bf16, M x K). Lane holds row M = lane&15;
// lanes 0-15 hold K = k0..k0+15, lanes 16-31 hold
// K = k0+16..k0+31 (matches load_b). rowp points at the row's K run.
__device__ inline v16bf load_a(const unsigned short* __restrict__ rowp, int k0, int lane) {
  int base = k0 + ((lane & 16) ? 16 : 0);
  union { v16bf v; v8bf h[2]; } u;
  u.h[0] = *(const v8bf*)(rowp + base);        // 16B aligned
  u.h[1] = *(const v8bf*)(rowp + base + 8);    // 16B aligned
  return u.v;
}

// B-fragment (32x16 bf16, K x N). Order used here (same as load_a): lane holds column N = lane&15,
// 16 contiguous K values (lanes 0-15: K=k0..k0+15, lanes 16-31: K=k0+16..k0+31).
// For our NT GEMMs this is one aligned 32-byte load from the operand row.
__device__ inline v16bf load_b(const unsigned short* __restrict__ p) {
  return *(const v16bf*)p;
}

__device__ inline v8f wmma_bf16(v16bf a, v16bf b, v8f c) {
  // (neg_a, A, neg_b, B, c_mod, C, reuse_a, reuse_b)
  return __builtin_amdgcn_wmma_f32_16x16x32_bf16(false, a, false, b, (short)0, c, false, false);
}

__device__ inline float sigmoidf_(float x) { return 1.0f / (1.0f + __expf(-x)); }

// ---------------- kernels ----------------

// fp32 -> bf16 conversion, 8 elements/thread/iter (16B stores), n must be %8
__global__ void cvt_f32_bf16_v8(const float* __restrict__ in,
                                unsigned short* __restrict__ out, int n8) {
  int stride = gridDim.x * blockDim.x;
  for (int i = blockIdx.x * blockDim.x + threadIdx.x; i < n8; i += stride) {
    v4f x0 = *(const v4f*)(in + (size_t)i * 8);
    v4f x1 = *(const v4f*)(in + (size_t)i * 8 + 4);
    v8us r;
#pragma unroll
    for (int j = 0; j < 4; ++j) { r[j] = f2bf(x0[j]); r[j + 4] = f2bf(x1[j]); }
    *(v8us*)(out + (size_t)i * 8) = r;
  }
}

// xg[m, n] = sum_k emb[token(m), k] * w_ih[n, k] + b_ih[n]
// m = b*TT + t (token gather fused into A-fragment loads). M=4096, N=1536, K=512.
// One wave computes a 32x64 strip; K-loop software-pipelined (load k+1 before
// consuming k). 24 N-strips * 128 M-tiles = 3072 wave jobs -> 384 blocks.
__global__ void __launch_bounds__(256) xg_gemm(
    const unsigned short* __restrict__ embbf,
    const unsigned short* __restrict__ wihbf,
    const float* __restrict__ b_ih,
    const int* __restrict__ target,
    float* __restrict__ xg) {
  int lane = threadIdx.x & 31;
  int waveJob = blockIdx.x * (blockDim.x >> 5) + (threadIdx.x >> 5);
  int n0 = (waveJob % 24) * 64;
  int m0 = (waveJob / 24) * 32;

  // Teacher-forced tokens for this lane's two A rows
  int mA0 = m0 + (lane & 15);
  int mA1 = mA0 + 16;
  int b0i = mA0 >> 7, t0i = mA0 & 127;
  int b1i = mA1 >> 7, t1i = mA1 & 127;
  int tok0 = (t0i == 0) ? 1 : target[b0i * TT + t0i - 1]; // START_TOKEN = 1
  int tok1 = (t1i == 0) ? 1 : target[b1i * TT + t1i - 1];
  const unsigned short* arow0 = embbf + (size_t)tok0 * HH;
  const unsigned short* arow1 = embbf + (size_t)tok1 * HH;
  const unsigned short* bbase = wihbf + (size_t)(n0 + (lane & 15)) * HH
                                      + ((lane & 16) ? 16 : 0);

  v8f acc[2][4] = {};

  v16bf a0 = load_a(arow0, 0, lane);
  v16bf a1 = load_a(arow1, 0, lane);
  v16bf b0 = load_b(bbase + (size_t)0 * 16 * HH);
  v16bf b1 = load_b(bbase + (size_t)1 * 16 * HH);
  v16bf b2 = load_b(bbase + (size_t)2 * 16 * HH);
  v16bf b3 = load_b(bbase + (size_t)3 * 16 * HH);

  for (int k0 = 0; k0 < HH - 32; k0 += 32) {
    int kn = k0 + 32;
    v16bf na0 = load_a(arow0, kn, lane);
    v16bf na1 = load_a(arow1, kn, lane);
    v16bf nb0 = load_b(bbase + (size_t)0 * 16 * HH + kn);
    v16bf nb1 = load_b(bbase + (size_t)1 * 16 * HH + kn);
    v16bf nb2 = load_b(bbase + (size_t)2 * 16 * HH + kn);
    v16bf nb3 = load_b(bbase + (size_t)3 * 16 * HH + kn);
    acc[0][0] = wmma_bf16(a0, b0, acc[0][0]);
    acc[1][0] = wmma_bf16(a1, b0, acc[1][0]);
    acc[0][1] = wmma_bf16(a0, b1, acc[0][1]);
    acc[1][1] = wmma_bf16(a1, b1, acc[1][1]);
    acc[0][2] = wmma_bf16(a0, b2, acc[0][2]);
    acc[1][2] = wmma_bf16(a1, b2, acc[1][2]);
    acc[0][3] = wmma_bf16(a0, b3, acc[0][3]);
    acc[1][3] = wmma_bf16(a1, b3, acc[1][3]);
    a0 = na0; a1 = na1; b0 = nb0; b1 = nb1; b2 = nb2; b3 = nb3;
  }
  // epilogue (last K block)
  acc[0][0] = wmma_bf16(a0, b0, acc[0][0]);
  acc[1][0] = wmma_bf16(a1, b0, acc[1][0]);
  acc[0][1] = wmma_bf16(a0, b1, acc[0][1]);
  acc[1][1] = wmma_bf16(a1, b1, acc[1][1]);
  acc[0][2] = wmma_bf16(a0, b2, acc[0][2]);
  acc[1][2] = wmma_bf16(a1, b2, acc[1][2]);
  acc[0][3] = wmma_bf16(a0, b3, acc[0][3]);
  acc[1][3] = wmma_bf16(a1, b3, acc[1][3]);

  int mrel = (lane & 16) ? 8 : 0;
#pragma unroll
  for (int mt = 0; mt < 2; ++mt) {
#pragma unroll
    for (int t = 0; t < 4; ++t) {
      int n = n0 + t * 16 + (lane & 15);
      float bias = b_ih[n];
#pragma unroll
      for (int j = 0; j < 8; ++j)
        xg[(size_t)(m0 + mt * 16 + mrel + j) * G3H + n] = acc[mt][t][j] + bias;
    }
  }
}

// One GRU step: hg = h_prev @ w_hh^T + b_hh, gate math, h_new + bf16 hs write.
// 8 blocks of 128 threads; block covers a 64-column strip; each wave owns 16
// columns across all 3 gates and both 16-row M tiles. B operand (w_hh rows)
// is double-buffered; A comes from LDS.
__global__ void __launch_bounds__(128) gru_step(
    const float* __restrict__ hin,
    float* __restrict__ hout,
    const unsigned short* __restrict__ whhbf,
    const float* __restrict__ xg,
    const float* __restrict__ b_hh,
    unsigned short* __restrict__ hsbf,
    int t) {
  __shared__ __align__(32) unsigned short hlds[BB * HH]; // 32KB bf16 h_prev
  for (int i = threadIdx.x; i < BB * HH; i += blockDim.x)
    hlds[i] = f2bf(hin[i]);
  __syncthreads();

  int lane = threadIdx.x & 31;
  int wave = threadIdx.x >> 5;
  int cN = blockIdx.x * 64 + wave * 16 + (lane & 15); // hidden-unit column

  const unsigned short* bp0 = whhbf + (size_t)(0 * HH + cN) * HH + ((lane & 16) ? 16 : 0);
  const unsigned short* bp1 = whhbf + (size_t)(1 * HH + cN) * HH + ((lane & 16) ? 16 : 0);
  const unsigned short* bp2 = whhbf + (size_t)(2 * HH + cN) * HH + ((lane & 16) ? 16 : 0);

  v8f acc[2][3] = {};
  v16bf b0 = load_b(bp0), b1 = load_b(bp1), b2 = load_b(bp2);

  for (int k0 = 0; k0 < HH - 32; k0 += 32) {
    int kn = k0 + 32;
    v16bf nb0 = load_b(bp0 + kn);
    v16bf nb1 = load_b(bp1 + kn);
    v16bf nb2 = load_b(bp2 + kn);
    v16bf a0 = load_a(hlds + (size_t)((lane & 15) + 0) * HH, k0, lane);
    v16bf a1 = load_a(hlds + (size_t)((lane & 15) + 16) * HH, k0, lane);
    acc[0][0] = wmma_bf16(a0, b0, acc[0][0]);
    acc[1][0] = wmma_bf16(a1, b0, acc[1][0]);
    acc[0][1] = wmma_bf16(a0, b1, acc[0][1]);
    acc[1][1] = wmma_bf16(a1, b1, acc[1][1]);
    acc[0][2] = wmma_bf16(a0, b2, acc[0][2]);
    acc[1][2] = wmma_bf16(a1, b2, acc[1][2]);
    b0 = nb0; b1 = nb1; b2 = nb2;
  }
  {
    int k0 = HH - 32;
    v16bf a0 = load_a(hlds + (size_t)((lane & 15) + 0) * HH, k0, lane);
    v16bf a1 = load_a(hlds + (size_t)((lane & 15) + 16) * HH, k0, lane);
    acc[0][0] = wmma_bf16(a0, b0, acc[0][0]);
    acc[1][0] = wmma_bf16(a1, b0, acc[1][0]);
    acc[0][1] = wmma_bf16(a0, b1, acc[0][1]);
    acc[1][1] = wmma_bf16(a1, b1, acc[1][1]);
    acc[0][2] = wmma_bf16(a0, b2, acc[0][2]);
    acc[1][2] = wmma_bf16(a1, b2, acc[1][2]);
  }

  float bhr = b_hh[0 * HH + cN];
  float bhz = b_hh[1 * HH + cN];
  float bhn = b_hh[2 * HH + cN];
#pragma unroll
  for (int mt = 0; mt < 2; ++mt) {
#pragma unroll
    for (int j = 0; j < 8; ++j) {
      int m = mt * 16 + ((lane & 16) ? 8 : 0) + j; // batch row
      size_t xoff = ((size_t)m * TT + t) * (size_t)G3H;
      float xr = xg[xoff + 0 * HH + cN];
      float xz = xg[xoff + 1 * HH + cN];
      float xn = xg[xoff + 2 * HH + cN];
      float r = sigmoidf_(xr + acc[mt][0][j] + bhr);
      float z = sigmoidf_(xz + acc[mt][1][j] + bhz);
      float n = tanhf(xn + r * (acc[mt][2][j] + bhn));
      float hp = hin[(size_t)m * HH + cN];
      float hv = (1.0f - z) * n + z * hp;
      hout[(size_t)m * HH + cN] = hv;
      hsbf[((size_t)t * BB + m) * HH + cN] = f2bf(hv); // hs is [T, B, H]
    }
  }
}

// logits[b, t, v] = sum_k hs[t, b, k] * emb[v, k] + b_out[v]
// M=4096 (t*B+b rows), N=32000, K=512. One wave -> 32x64 strip, K-loop
// software-pipelined. 500 N-strips * 128 M-tiles = 64000 jobs -> 8000 blocks.
__global__ void __launch_bounds__(256) logits_gemm(
    const unsigned short* __restrict__ hsbf,
    const unsigned short* __restrict__ embbf,
    const float* __restrict__ b_out,
    float* __restrict__ out) {
  int lane = threadIdx.x & 31;
  int waveJob = blockIdx.x * (blockDim.x >> 5) + (threadIdx.x >> 5);
  int n0 = (waveJob % 500) * 64;
  int m0 = (waveJob / 500) * 32;

  const unsigned short* arow0 = hsbf + (size_t)(m0 + (lane & 15)) * HH;
  const unsigned short* arow1 = arow0 + (size_t)16 * HH;
  const unsigned short* bbase = embbf + (size_t)(n0 + (lane & 15)) * HH
                                      + ((lane & 16) ? 16 : 0);

  v8f acc[2][4] = {};

  v16bf a0 = load_a(arow0, 0, lane);
  v16bf a1 = load_a(arow1, 0, lane);
  v16bf b0 = load_b(bbase + (size_t)0 * 16 * HH);
  v16bf b1 = load_b(bbase + (size_t)1 * 16 * HH);
  v16bf b2 = load_b(bbase + (size_t)2 * 16 * HH);
  v16bf b3 = load_b(bbase + (size_t)3 * 16 * HH);

  for (int k0 = 0; k0 < HH - 32; k0 += 32) {
    int kn = k0 + 32;
    v16bf na0 = load_a(arow0, kn, lane);
    v16bf na1 = load_a(arow1, kn, lane);
    v16bf nb0 = load_b(bbase + (size_t)0 * 16 * HH + kn);
    v16bf nb1 = load_b(bbase + (size_t)1 * 16 * HH + kn);
    v16bf nb2 = load_b(bbase + (size_t)2 * 16 * HH + kn);
    v16bf nb3 = load_b(bbase + (size_t)3 * 16 * HH + kn);
    acc[0][0] = wmma_bf16(a0, b0, acc[0][0]);
    acc[1][0] = wmma_bf16(a1, b0, acc[1][0]);
    acc[0][1] = wmma_bf16(a0, b1, acc[0][1]);
    acc[1][1] = wmma_bf16(a1, b1, acc[1][1]);
    acc[0][2] = wmma_bf16(a0, b2, acc[0][2]);
    acc[1][2] = wmma_bf16(a1, b2, acc[1][2]);
    acc[0][3] = wmma_bf16(a0, b3, acc[0][3]);
    acc[1][3] = wmma_bf16(a1, b3, acc[1][3]);
    a0 = na0; a1 = na1; b0 = nb0; b1 = nb1; b2 = nb2; b3 = nb3;
  }
  // epilogue (last K block)
  acc[0][0] = wmma_bf16(a0, b0, acc[0][0]);
  acc[1][0] = wmma_bf16(a1, b0, acc[1][0]);
  acc[0][1] = wmma_bf16(a0, b1, acc[0][1]);
  acc[1][1] = wmma_bf16(a1, b1, acc[1][1]);
  acc[0][2] = wmma_bf16(a0, b2, acc[0][2]);
  acc[1][2] = wmma_bf16(a1, b2, acc[1][2]);
  acc[0][3] = wmma_bf16(a0, b3, acc[0][3]);
  acc[1][3] = wmma_bf16(a1, b3, acc[1][3]);

  int mrel = (lane & 16) ? 8 : 0;
#pragma unroll
  for (int mt = 0; mt < 2; ++mt) {
#pragma unroll
    for (int tt = 0; tt < 4; ++tt) {
      int n = n0 + tt * 16 + (lane & 15);
      float bias = b_out[n];
#pragma unroll
      for (int j = 0; j < 8; ++j) {
        int m = m0 + mt * 16 + mrel + j; // hs row = t*BB + b
        int tpos = m >> 5;
        int bidx = m & 31;
        out[((size_t)bidx * TT + tpos) * (size_t)VV + n] = acc[mt][tt][j] + bias;
      }
    }
  }
}

// ---------------- host launcher ----------------

extern "C" void kernel_launch(void* const* d_in, const int* in_sizes, int n_in,
                              void* d_out, int out_size, void* d_ws, size_t ws_size,
                              hipStream_t stream) {
  (void)in_sizes; (void)n_in; (void)out_size; (void)ws_size;
  const float* emb   = (const float*)d_in[0];
  const float* wih   = (const float*)d_in[1];
  const float* whh   = (const float*)d_in[2];
  const float* b_ih  = (const float*)d_in[3];
  const float* b_hh  = (const float*)d_in[4];
  const float* b_out = (const float*)d_in[5];
  const float* enc   = (const float*)d_in[6]; // [1, B, H] == [B, H]
  const int*   tgt   = (const int*)d_in[7];
  float* out = (float*)d_out;

  char* ws = (char*)d_ws;
  size_t off = 0;
  auto alloc = [&](size_t bytes) -> void* {
    void* p = ws + off;
    off += (bytes + 255) & ~(size_t)255;
    return p;
  };
  unsigned short* embbf = (unsigned short*)alloc((size_t)VV * HH * 2);      // 32.8 MB
  unsigned short* wihbf = (unsigned short*)alloc((size_t)G3H * HH * 2);     // 1.5 MB
  unsigned short* whhbf = (unsigned short*)alloc((size_t)G3H * HH * 2);     // 1.5 MB
  float*          xg    = (float*)alloc((size_t)BB * TT * G3H * 4);         // 25.2 MB
  unsigned short* hsbf  = (unsigned short*)alloc((size_t)TT * BB * HH * 2); // 4.2 MB
  float*          hbuf  = (float*)alloc((size_t)2 * BB * HH * 4);           // 128 KB

  // 1) bf16 conversions of the reused weight matrices (vectorized, n % 8 == 0)
  cvt_f32_bf16_v8<<<2048, 256, 0, stream>>>(emb, embbf, (VV * HH) / 8);
  cvt_f32_bf16_v8<<<96, 256, 0, stream>>>(wih, wihbf, (G3H * HH) / 8);
  cvt_f32_bf16_v8<<<96, 256, 0, stream>>>(whh, whhbf, (G3H * HH) / 8);

  // 2) input-side gate preactivations (token gather fused)
  xg_gemm<<<(128 * 24) / 8, 256, 0, stream>>>(embbf, wihbf, b_ih, tgt, xg);

  // 3) sequential GRU scan: 128 stream-ordered steps, double-buffered h
  for (int t = 0; t < TT; ++t) {
    const float* hin = (t == 0) ? enc : (hbuf + (size_t)(t & 1) * BB * HH);
    float* hout = hbuf + (size_t)((t + 1) & 1) * BB * HH;
    gru_step<<<HH / 64, 128, 0, stream>>>(hin, hout, whhbf, xg, b_hh, hsbf, t);
  }

  // 4) tied-weight output projection (dominant GEMM: 134 GFLOP, 524 MB store)
  logits_gemm<<<(128 * 500) / 8, 256, 0, stream>>>(hsbf, embbf, b_out, out);
}
